// LocalGNN_2224793059852
// MI455X (gfx1250) — matrix operand order repaired and checked by reading, not compiled, not run on hardware
//
#include <hip/hip_runtime.h>

#define NN 100000
#define NE 1600000
#define DHID 128
#define DOUT 64
#define BN_EPS 1e-5f

typedef __attribute__((ext_vector_type(8)))  unsigned short v8u;
typedef __attribute__((ext_vector_type(16))) unsigned short v16u;
typedef __attribute__((ext_vector_type(16))) __bf16         v16bf;
typedef __attribute__((ext_vector_type(8)))  float          v8f;

__device__ __forceinline__ unsigned short f2bf(float f) {
  unsigned u = __float_as_uint(f);
  u += 0x7FFFu + ((u >> 16) & 1u);      // round-to-nearest-even
  return (unsigned short)(u >> 16);
}

// ---------------- degree / dinv ----------------
__global__ void k_zero(float* __restrict__ p, int n) {
  int i = blockIdx.x * blockDim.x + threadIdx.x;
  if (i < n) p[i] = 0.0f;
}

__global__ void k_deg(const int* __restrict__ dst, float* __restrict__ deg, int e) {
  int i = blockIdx.x * blockDim.x + threadIdx.x;
  if (i < e) atomicAdd(&deg[dst[i]], 1.0f);
}

__global__ void k_dinv(const float* __restrict__ deg, float* __restrict__ dinv, int n) {
  int i = blockIdx.x * blockDim.x + threadIdx.x;
  if (i < n) dinv[i] = rsqrtf(deg[i] + 1.0f);
}

// ---------------- f32 -> bf16 feature convert ----------------
__global__ void k_f32_to_bf16(const float* __restrict__ x, unsigned short* __restrict__ xb, int n) {
  int i = blockIdx.x * blockDim.x + threadIdx.x;
  if (i < n) xb[i] = f2bf(x[i]);
}

// ---------------- pack W[k][n] (f32 row-major) into WMMA B-lane order (bf16) ----
// Layout: Wp[((nt*4 + ks)*32 + lane)*16 + j], lane: n = lane%16, halfB = lane/16;
// within lane: element j -> K = ks*32 + (j/8)*16 + halfB*8 + (j%8)
__global__ void k_packW(const float* __restrict__ W, unsigned short* __restrict__ Wp, int dout) {
  int p = blockIdx.x * blockDim.x + threadIdx.x;
  int total = (dout >> 4) * 4 * 32 * 16;
  if (p >= total) return;
  int j    = p & 15;
  int lane = (p >> 4) & 31;
  int ks   = (p >> 9) & 3;
  int nt   = p >> 11;
  int n  = lane & 15;
  int hb = lane >> 4;
  int k  = ks * 32 + (j >> 3) * 16 + hb * 8 + (j & 7);
  Wp[p] = f2bf(W[k * dout + nt * 16 + n]);
}

// ---------------- GEMM: H[N][dout] = Xb[N][128] @ W, via bf16 WMMA ----------------
// one wave per 16x16 output tile; blockDim.x = 32 * (dout/16); blockIdx.x = m-tile
__global__ void k_gemm(const unsigned short* __restrict__ Xb,
                       const unsigned short* __restrict__ Wp,
                       float* __restrict__ H, int dout) {
  int lane = threadIdx.x & 31;
  int wave = threadIdx.x >> 5;            // n-tile index
  int mt   = blockIdx.x;
  int m    = lane & 15;
  int half = lane >> 4;
  const unsigned short* xrow = Xb + (size_t)(mt * 16 + m) * DHID;

  v8f c = {};
#pragma unroll
  for (int ks = 0; ks < 4; ++ks) {
    int kb = ks * 32;
    // A 16x32 bf16 layout (ISA 7.12.2): VGPR0-3 <- K = half*8..+7, VGPR4-7 <- K = 16+half*8..+7
    v8u lo = *(const v8u*)(xrow + kb + half * 8);
    v8u hi = *(const v8u*)(xrow + kb + 16 + half * 8);
    v16u av = __builtin_shufflevector(lo, hi, 0, 1, 2, 3, 4, 5, 6, 7,
                                              8, 9, 10, 11, 12, 13, 14, 15);
    v16u bv = *(const v16u*)(Wp + (((wave * 4 + ks) * 32 + lane) << 4));
    c = __builtin_amdgcn_wmma_f32_16x16x32_bf16(
        false, __builtin_bit_cast(v16bf, av),
        false, __builtin_bit_cast(v16bf, bv),
        (short)0, c, false, false);
  }
  // D 16x16 f32 layout: VGPR r -> row (r + half*8), col = lane%16
#pragma unroll
  for (int r = 0; r < 8; ++r) {
    H[(size_t)(mt * 16 + r + half * 8) * dout + wave * 16 + m] = c[r];
  }
}

// ---------------- AGG = dinv^2 * H + b  (self loop + bias) ----------------
__global__ void k_self_init(const float* __restrict__ H, const float* __restrict__ dinv,
                            const float* __restrict__ b, float* __restrict__ AGG,
                            int shift, int total) {
  int i = blockIdx.x * blockDim.x + threadIdx.x;
  if (i >= total) return;
  int node = i >> shift;
  int f    = i & ((1 << shift) - 1);
  float di = dinv[node];
  AGG[i] = di * di * H[i] + b[f];
}

// ---------------- edge scatter: AGG[dst] += dinv[src]*dinv[dst] * H[src] ----------
__global__ void k_scatter(const int* __restrict__ src, const int* __restrict__ dst,
                          const float* __restrict__ dinv, const float4* __restrict__ H4,
                          float* __restrict__ AGG, int dout, int cshift, int total) {
  int t = blockIdx.x * blockDim.x + threadIdx.x;
  if (t >= total) return;
  int chunks = dout >> 2;
  int e = t >> cshift;
  int c = t & (chunks - 1);
  int s = src[e];
  int d = dst[e];
  float w = dinv[s] * dinv[d];
  float4 h = H4[(size_t)s * chunks + c];
  float* base = AGG + (size_t)d * dout + (c << 2);
  atomicAdd(base + 0, w * h.x);
  atomicAdd(base + 1, w * h.y);
  atomicAdd(base + 2, w * h.z);
  atomicAdd(base + 3, w * h.w);
}

// ---------------- BN(eval) + ReLU + bf16 convert (feeds next GEMM) --------------
__global__ void k_bn_relu_bf16(const float* __restrict__ AGG,
                               const float* __restrict__ gamma, const float* __restrict__ beta,
                               const float* __restrict__ mean,  const float* __restrict__ var,
                               unsigned short* __restrict__ Xb, int total) {
  int i = blockIdx.x * blockDim.x + threadIdx.x;
  if (i >= total) return;
  int f = i & (DHID - 1);
  float v = (AGG[i] - mean[f]) * rsqrtf(var[f] + BN_EPS) * gamma[f] + beta[f];
  Xb[i] = f2bf(v > 0.0f ? v : 0.0f);
}

// ---------------- final ReLU ----------------
__global__ void k_relu(const float* __restrict__ A, float* __restrict__ out, int total) {
  int i = blockIdx.x * blockDim.x + threadIdx.x;
  if (i >= total) return;
  float v = A[i];
  out[i] = v > 0.0f ? v : 0.0f;
}

extern "C" void kernel_launch(void* const* d_in, const int* in_sizes, int n_in,
                              void* d_out, int out_size, void* d_ws, size_t ws_size,
                              hipStream_t stream) {
  const float* x   = (const float*)d_in[0];
  const int*   ei  = (const int*)d_in[1];
  const float* W1  = (const float*)d_in[2];
  const float* b1  = (const float*)d_in[3];
  const float* W2  = (const float*)d_in[4];
  const float* b2  = (const float*)d_in[5];
  const float* W3  = (const float*)d_in[6];
  const float* b3  = (const float*)d_in[7];
  const float* g1  = (const float*)d_in[8];
  const float* be1 = (const float*)d_in[9];
  const float* m1  = (const float*)d_in[10];
  const float* v1  = (const float*)d_in[11];
  const float* g2  = (const float*)d_in[12];
  const float* be2 = (const float*)d_in[13];
  const float* m2  = (const float*)d_in[14];
  const float* v2  = (const float*)d_in[15];
  float* out = (float*)d_out;

  const int* esrc = ei;
  const int* edst = ei + NE;

  // workspace carve-out (256B aligned slabs)
  char* ws = (char*)d_ws;
  size_t off = 0;
  auto take = [&](size_t bytes) -> char* {
    char* r = ws + off;
    off = (off + bytes + 255) & ~(size_t)255;
    return r;
  };
  float*          deg  = (float*)take((size_t)NN * 4);
  float*          dinv = (float*)take((size_t)NN * 4);
  unsigned short* Xb   = (unsigned short*)take((size_t)NN * DHID * 2);
  float*          H    = (float*)take((size_t)NN * DHID * 4);
  float*          AGG  = (float*)take((size_t)NN * DHID * 4);
  unsigned short* Wp   = (unsigned short*)take((size_t)8 * 4 * 32 * 16 * 2);

  const int T = 256;
  auto nb = [](int n, int t) { return (n + t - 1) / t; };

  // degrees
  k_zero<<<nb(NN, T), T, 0, stream>>>(deg, NN);
  k_deg<<<nb(NE, T), T, 0, stream>>>(edst, deg, NE);
  k_dinv<<<nb(NN, T), T, 0, stream>>>(deg, dinv, NN);

  // input features -> bf16
  k_f32_to_bf16<<<nb(NN * DHID, T), T, 0, stream>>>(x, Xb, NN * DHID);

  // ---- layer 1: dout = 128 ----
  k_packW<<<nb(8 * 4 * 32 * 16, T), T, 0, stream>>>(W1, Wp, DHID);
  k_gemm<<<NN / 16, 256, 0, stream>>>(Xb, Wp, H, DHID);
  k_self_init<<<nb(NN * DHID, T), T, 0, stream>>>(H, dinv, b1, AGG, 7, NN * DHID);
  k_scatter<<<nb(NE * 32, T), T, 0, stream>>>(esrc, edst, dinv, (const float4*)H, AGG,
                                              DHID, 5, NE * 32);
  k_bn_relu_bf16<<<nb(NN * DHID, T), T, 0, stream>>>(AGG, g1, be1, m1, v1, Xb, NN * DHID);

  // ---- layer 2: dout = 128 ----
  k_packW<<<nb(8 * 4 * 32 * 16, T), T, 0, stream>>>(W2, Wp, DHID);
  k_gemm<<<NN / 16, 256, 0, stream>>>(Xb, Wp, H, DHID);
  k_self_init<<<nb(NN * DHID, T), T, 0, stream>>>(H, dinv, b2, AGG, 7, NN * DHID);
  k_scatter<<<nb(NE * 32, T), T, 0, stream>>>(esrc, edst, dinv, (const float4*)H, AGG,
                                              DHID, 5, NE * 32);
  k_bn_relu_bf16<<<nb(NN * DHID, T), T, 0, stream>>>(AGG, g2, be2, m2, v2, Xb, NN * DHID);

  // ---- layer 3: dout = 64 ----
  k_packW<<<nb(4 * 4 * 32 * 16, T), T, 0, stream>>>(W3, Wp, DOUT);
  k_gemm<<<NN / 16, 128, 0, stream>>>(Xb, Wp, H, DOUT);
  k_self_init<<<nb(NN * DOUT, T), T, 0, stream>>>(H, dinv, b3, AGG, 6, NN * DOUT);
  k_scatter<<<nb(NE * 16, T), T, 0, stream>>>(esrc, edst, dinv, (const float4*)H, AGG,
                                              DOUT, 4, NE * 16);
  k_relu<<<nb(NN * DOUT, T), T, 0, stream>>>(AGG, out, NN * DOUT);
}
